// DockingSurrogateModel_57251914056304
// MI455X (gfx1250) — compile-verified, matrix operand order checked by reading, not run on hardware
//
#include <hip/hip_runtime.h>

// ---------------------------------------------------------------------------
// Types / helpers
// ---------------------------------------------------------------------------
typedef __bf16 bf16_t;
typedef __attribute__((ext_vector_type(16))) __bf16 v16bf;
typedef __attribute__((ext_vector_type(8)))  __bf16 v8bf;
typedef __attribute__((ext_vector_type(8)))  float  v8f;

#define NNODES 50000
#define NEDGES 400000
#define NB     128
#define HDIM   128

__device__ __forceinline__ bf16_t f2bf(float f) {
    unsigned u = __builtin_bit_cast(unsigned, f);
    unsigned r = u + 0x7FFFu + ((u >> 16) & 1u);
    unsigned short h = (unsigned short)(r >> 16);
    return __builtin_bit_cast(bf16_t, h);
}
__device__ __forceinline__ float bf2f(bf16_t b) {
    unsigned short h = __builtin_bit_cast(unsigned short, b);
    unsigned u = ((unsigned)h) << 16;
    return __builtin_bit_cast(float, u);
}
__device__ __forceinline__ unsigned flipf(float f) {
    unsigned u = __builtin_bit_cast(unsigned, f);
    return (u & 0x80000000u) ? ~u : (u | 0x80000000u);
}
__device__ __forceinline__ float unflipf(unsigned u) {
    unsigned v = (u & 0x80000000u) ? (u & 0x7FFFFFFFu) : ~u;
    return __builtin_bit_cast(float, v);
}
// per-lane K index inside a 32-wide k-step for 16-bit WMMA operands
// lane<16 : elems 0..7 -> kk 0..7,   elems 8..15 -> kk 16..23
// lane>=16: elems 0..7 -> kk 8..15,  elems 8..15 -> kk 24..31

// ---------------------------------------------------------------------------
// Weight pre-pack: fp32 W[K][N] -> bf16 fragment-major (frag, lane, elem)
// frag = kstep*(N/16)+nt ; each lane owns 16 contiguous bf16 (32B)
// ---------------------------------------------------------------------------
__global__ void prepack_kernel(const float* __restrict__ W, bf16_t* __restrict__ P,
                               int K, int N, int KP) {
    int NT = N >> 4;
    long total = (long)(KP >> 5) * NT * 512;
    for (long idx = (long)blockIdx.x * blockDim.x + threadIdx.x; idx < total;
         idx += (long)gridDim.x * blockDim.x) {
        int e    = (int)(idx & 15);
        int lane = (int)((idx >> 4) & 31);
        int frag = (int)(idx >> 9);
        int ks = frag / NT, nt = frag % NT;
        int half = lane >> 4;
        int kk = half ? ((e < 8) ? (8 + e) : (16 + e))
                      : ((e < 8) ? e : (8 + e));
        int k = ks * 32 + kk;
        int n = nt * 16 + (lane & 15);
        float v = (k < K && n < N) ? W[(size_t)k * N + n] : 0.0f;
        P[idx] = f2bf(v);
    }
}

// ---------------------------------------------------------------------------
// Generic fused linear: out = act(A @ W + b), A fp32 [M,lda], W packed bf16.
// One wave per 16-row tile; NTILES*16 output columns.
// ---------------------------------------------------------------------------
template<int NTILES, int ACT>
__launch_bounds__(128)
__global__ void wmma_linear_kernel(const float* __restrict__ A, int lda, int M,
                                   int K, int KP,
                                   const bf16_t* __restrict__ Wp,
                                   const float* __restrict__ bias,
                                   float* __restrict__ outf,
                                   bf16_t* __restrict__ outh, int ldo) {
    const int lane = threadIdx.x & 31;
    const int wave = threadIdx.x >> 5;
    const int half = lane >> 4;
    const int lm   = lane & 15;
    const int mtiles = (M + 15) >> 4;
    const int ksteps = KP >> 5;
    for (int t = blockIdx.x * 4 + wave; t < mtiles; t += gridDim.x * 4) {
        int row = t * 16 + lm; if (row >= M) row = M - 1;
        const float* arow = A + (size_t)row * lda;
        v8f acc[NTILES];
        #pragma unroll
        for (int nt = 0; nt < NTILES; ++nt) acc[nt] = v8f{};
        for (int ks = 0; ks < ksteps; ++ks) {
            int kb0 = ks * 32 + half * 8;
            int kb1 = kb0 + 16;
            v16bf a;
            #pragma unroll
            for (int e = 0; e < 8; ++e) {
                int k0 = kb0 + e, k1 = kb1 + e;
                a[e]     = (k0 < K) ? f2bf(arow[k0]) : f2bf(0.f);
                a[e + 8] = (k1 < K) ? f2bf(arow[k1]) : f2bf(0.f);
            }
            const bf16_t* wp = Wp + (size_t)ks * NTILES * 512 + lane * 16;
            #pragma unroll
            for (int nt = 0; nt < NTILES; ++nt) {
                v16bf b = *(const v16bf*)(wp + nt * 512);
                acc[nt] = __builtin_amdgcn_wmma_f32_16x16x32_bf16(
                    false, a, false, b, (short)0, acc[nt], false, false);
            }
        }
        #pragma unroll
        for (int nt = 0; nt < NTILES; ++nt) {
            int n = nt * 16 + lm;
            float bn = bias ? bias[n] : 0.f;
            #pragma unroll
            for (int r = 0; r < 8; ++r) {
                int m = t * 16 + half * 8 + r;
                if (m < M) {
                    float v = acc[nt][r] + bn;
                    if (ACT) v = fmaxf(v, 0.f);
                    if (outf) outf[(size_t)m * ldo + n] = v;
                    if (outh) outh[(size_t)m * ldo + n] = f2bf(v);
                }
            }
        }
    }
}

// ---------------------------------------------------------------------------
// Fused message kernel, 2 edge-tiles (32 edges) per wave so each LDS B-frag
// load feeds two WMMAs (LDS bandwidth is the limiter, not the matrix pipe).
// h = relu([x[row],x[col],ea] @ W1 + b1) @ W2 + b2, plus attention logit.
// Dynamic LDS = 49152*2 + 16384*2 + 4*2*2176*2 + 512 = 166400 bytes.
// ---------------------------------------------------------------------------
__launch_bounds__(128)
__global__ void message_kernel(const bf16_t* __restrict__ xh,
                               const bf16_t* __restrict__ eah,
                               const int* __restrict__ ei,
                               const bf16_t* __restrict__ w1p,
                               const bf16_t* __restrict__ w2p,
                               const float* __restrict__ b1,
                               const float* __restrict__ b2,
                               const float* __restrict__ aw,
                               const float* __restrict__ ab,
                               bf16_t* __restrict__ hbuf,
                               float* __restrict__ logits, int E) {
    extern __shared__ char smem[];
    bf16_t* s_w1 = (bf16_t*)smem;            // 384*128
    bf16_t* s_w2 = s_w1 + 49152;             // 128*128
    bf16_t* s_ht_all = s_w2 + 16384;         // 4 waves * 2 tiles * 16*136
    float*  s_aw = (float*)(s_ht_all + 4 * 2 * 2176);
    const int tid = threadIdx.x;
    {
        const uint4* s1 = (const uint4*)w1p; uint4* d1 = (uint4*)s_w1;
        for (int i = tid; i < 98304 / 16; i += blockDim.x) d1[i] = s1[i];
        const uint4* s2 = (const uint4*)w2p; uint4* d2 = (uint4*)s_w2;
        for (int i = tid; i < 32768 / 16; i += blockDim.x) d2[i] = s2[i];
        if (tid < 128) s_aw[tid] = aw[tid];
    }
    __syncthreads();
    const int lane = tid & 31, wave = tid >> 5;
    const int half = lane >> 4, lm = lane & 15;
    bf16_t* ht0 = s_ht_all + wave * 2 * 2176;
    bf16_t* ht1 = ht0 + 2176;
    const float abv = ab[0];
    float bn1[8], bn2[8];
    #pragma unroll
    for (int nt = 0; nt < 8; ++nt) { bn1[nt] = b1[nt*16+lm]; bn2[nt] = b2[nt*16+lm]; }
    const int npair = E >> 5;   // 400000/32 = 12500 exact
    for (int tp = blockIdx.x * 4 + wave; tp < npair; tp += gridDim.x * 4) {
        const long t0 = (long)tp * 2, t1 = t0 + 1;
        const long e0 = t0 * 16 + lm, e1 = t1 * 16 + lm;
        const int row0 = ei[e0], col0 = ei[(long)E + e0];
        const int row1 = ei[e1], col1 = ei[(long)E + e1];
        const bf16_t* sxr0 = xh  + (size_t)row0 * 128;
        const bf16_t* sxc0 = xh  + (size_t)col0 * 128;
        const bf16_t* sea0 = eah + (size_t)e0   * 128;
        const bf16_t* sxr1 = xh  + (size_t)row1 * 128;
        const bf16_t* sxc1 = xh  + (size_t)col1 * 128;
        const bf16_t* sea1 = eah + (size_t)e1   * 128;
        v8f acc0[8], acc1[8];
        #pragma unroll
        for (int nt = 0; nt < 8; ++nt) { acc0[nt] = v8f{}; acc1[nt] = v8f{}; }
        #pragma unroll
        for (int ks = 0; ks < 12; ++ks) {
            int k0 = ks * 32 + half * 8;
            int k1 = k0 + 16;
            const bf16_t* p00 = (k0 < 128) ? sxr0 + k0 : ((k0 < 256) ? sxc0 + (k0-128) : sea0 + (k0-256));
            const bf16_t* p01 = (k1 < 128) ? sxr0 + k1 : ((k1 < 256) ? sxc0 + (k1-128) : sea0 + (k1-256));
            const bf16_t* p10 = (k0 < 128) ? sxr1 + k0 : ((k0 < 256) ? sxc1 + (k0-128) : sea1 + (k0-256));
            const bf16_t* p11 = (k1 < 128) ? sxr1 + k1 : ((k1 < 256) ? sxc1 + (k1-128) : sea1 + (k1-256));
            v8bf c00 = *(const v8bf*)p00;
            v8bf c01 = *(const v8bf*)p01;
            v8bf c10 = *(const v8bf*)p10;
            v8bf c11 = *(const v8bf*)p11;
            v16bf a0, a1;
            #pragma unroll
            for (int i = 0; i < 8; ++i) {
                a0[i] = c00[i]; a0[i+8] = c01[i];
                a1[i] = c10[i]; a1[i+8] = c11[i];
            }
            const bf16_t* wp = s_w1 + (size_t)ks * 8 * 512 + lane * 16;
            #pragma unroll
            for (int nt = 0; nt < 8; ++nt) {
                v16bf b = *(const v16bf*)(wp + nt * 512);
                acc0[nt] = __builtin_amdgcn_wmma_f32_16x16x32_bf16(
                    false, a0, false, b, (short)0, acc0[nt], false, false);
                acc1[nt] = __builtin_amdgcn_wmma_f32_16x16x32_bf16(
                    false, a1, false, b, (short)0, acc1[nt], false, false);
            }
        }
        // relu(h1) -> LDS tiles (stride 136 bf16 to dodge bank conflicts)
        #pragma unroll
        for (int nt = 0; nt < 8; ++nt) {
            int n = nt * 16 + lm;
            #pragma unroll
            for (int r = 0; r < 8; ++r) {
                int m = half * 8 + r;
                ht0[m * 136 + n] = f2bf(fmaxf(acc0[nt][r] + bn1[nt], 0.f));
                ht1[m * 136 + n] = f2bf(fmaxf(acc1[nt][r] + bn1[nt], 0.f));
            }
        }
        // GEMM2: h2 = h1 @ W2 + b2 (B shared across both tiles again)
        v8f a20[8], a21[8];
        #pragma unroll
        for (int nt = 0; nt < 8; ++nt) { a20[nt] = v8f{}; a21[nt] = v8f{}; }
        #pragma unroll
        for (int ks = 0; ks < 4; ++ks) {
            int k0 = ks * 32 + half * 8;
            v8bf c00 = *(const v8bf*)(ht0 + lm * 136 + k0);
            v8bf c01 = *(const v8bf*)(ht0 + lm * 136 + k0 + 16);
            v8bf c10 = *(const v8bf*)(ht1 + lm * 136 + k0);
            v8bf c11 = *(const v8bf*)(ht1 + lm * 136 + k0 + 16);
            v16bf a0, a1;
            #pragma unroll
            for (int i = 0; i < 8; ++i) {
                a0[i] = c00[i]; a0[i+8] = c01[i];
                a1[i] = c10[i]; a1[i+8] = c11[i];
            }
            const bf16_t* wp = s_w2 + (size_t)ks * 8 * 512 + lane * 16;
            #pragma unroll
            for (int nt = 0; nt < 8; ++nt) {
                v16bf b = *(const v16bf*)(wp + nt * 512);
                a20[nt] = __builtin_amdgcn_wmma_f32_16x16x32_bf16(
                    false, a0, false, b, (short)0, a20[nt], false, false);
                a21[nt] = __builtin_amdgcn_wmma_f32_16x16x32_bf16(
                    false, a1, false, b, (short)0, a21[nt], false, false);
            }
        }
        #pragma unroll
        for (int nt = 0; nt < 8; ++nt) {
            int n = nt * 16 + lm;
            #pragma unroll
            for (int r = 0; r < 8; ++r) {
                int m = half * 8 + r;
                ht0[m * 136 + n] = f2bf(a20[nt][r] + bn2[nt]);
                ht1[m * 136 + n] = f2bf(a21[nt][r] + bn2[nt]);
            }
        }
        // copy both 16x128 tiles to global hbuf: one full row (256B) per lane
        {
            const bf16_t* s = (half ? ht1 : ht0) + lm * 136;
            bf16_t* d = hbuf + ((size_t)(half ? t1 : t0) * 16 + lm) * 128;
            #pragma unroll
            for (int j = 0; j < 16; ++j)
                *(v8bf*)(d + j * 8) = *(const v8bf*)(s + j * 8);
        }
        // attention logits: lanes 0-15 -> tile0 edges, lanes 16-31 -> tile1
        {
            const bf16_t* htl = half ? ht1 : ht0;
            float s = 0.f;
            for (int n = 0; n < 128; ++n) s += bf2f(htl[lm * 136 + n]) * s_aw[n];
            logits[(size_t)(half ? t1 : t0) * 16 + lm] = s + abv;
        }
    }
}

// ---------------------------------------------------------------------------
// Fused update kernel, 2 node-tiles per wave (same B-amortization):
// x = relu( relu([x,agg]@U1+c1)@U2+c2 + x )  (in place)
// Dynamic LDS = 32768*2 + 16384*2 + 4*2*2176*2 = 133120 bytes.
// ---------------------------------------------------------------------------
__launch_bounds__(128)
__global__ void update_kernel(float* __restrict__ xf, bf16_t* __restrict__ xh,
                              const float* __restrict__ agg,
                              const bf16_t* __restrict__ w1p,
                              const bf16_t* __restrict__ w2p,
                              const float* __restrict__ b1,
                              const float* __restrict__ b2, int M) {
    extern __shared__ char smem[];
    bf16_t* s_w1 = (bf16_t*)smem;            // 256*128
    bf16_t* s_w2 = s_w1 + 32768;             // 128*128
    bf16_t* s_ht_all = s_w2 + 16384;
    const int tid = threadIdx.x;
    {
        const uint4* s1 = (const uint4*)w1p; uint4* d1 = (uint4*)s_w1;
        for (int i = tid; i < 65536 / 16; i += blockDim.x) d1[i] = s1[i];
        const uint4* s2 = (const uint4*)w2p; uint4* d2 = (uint4*)s_w2;
        for (int i = tid; i < 32768 / 16; i += blockDim.x) d2[i] = s2[i];
    }
    __syncthreads();
    const int lane = tid & 31, wave = tid >> 5;
    const int half = lane >> 4, lm = lane & 15;
    bf16_t* ht0 = s_ht_all + wave * 2 * 2176;
    bf16_t* ht1 = ht0 + 2176;
    float bn1[8], bn2[8];
    #pragma unroll
    for (int nt = 0; nt < 8; ++nt) { bn1[nt] = b1[nt*16+lm]; bn2[nt] = b2[nt*16+lm]; }
    const int mtiles = M >> 4;              // 3125 (odd)
    const int npair = (mtiles + 1) >> 1;    // 1563
    for (int tp = blockIdx.x * 4 + wave; tp < npair; tp += gridDim.x * 4) {
        const int t0 = tp * 2;
        const int t1 = t0 + 1;
        const bool has1 = (t1 < mtiles);          // wave-uniform tail guard
        const int t1c = has1 ? t1 : t0;
        const int row0 = t0 * 16 + lm;
        const int row1 = t1c * 16 + lm;
        const bf16_t* xrow0 = xh + (size_t)row0 * 128;
        const float*  arow0 = agg + (size_t)row0 * 128;
        const bf16_t* xrow1 = xh + (size_t)row1 * 128;
        const float*  arow1 = agg + (size_t)row1 * 128;
        v8f acc0[8], acc1[8];
        #pragma unroll
        for (int nt = 0; nt < 8; ++nt) { acc0[nt] = v8f{}; acc1[nt] = v8f{}; }
        #pragma unroll
        for (int ks = 0; ks < 8; ++ks) {
            int k0 = ks * 32 + half * 8;
            int k1 = k0 + 16;
            v8bf c00, c01, c10, c11;
            if (k0 < 128) { c00 = *(const v8bf*)(xrow0 + k0); c10 = *(const v8bf*)(xrow1 + k0); }
            else {
                const float4* f0 = (const float4*)(arow0 + (k0 - 128));
                const float4* f1 = (const float4*)(arow1 + (k0 - 128));
                float4 u0 = f0[0], u1 = f0[1], w0 = f1[0], w1 = f1[1];
                c00[0]=f2bf(u0.x); c00[1]=f2bf(u0.y); c00[2]=f2bf(u0.z); c00[3]=f2bf(u0.w);
                c00[4]=f2bf(u1.x); c00[5]=f2bf(u1.y); c00[6]=f2bf(u1.z); c00[7]=f2bf(u1.w);
                c10[0]=f2bf(w0.x); c10[1]=f2bf(w0.y); c10[2]=f2bf(w0.z); c10[3]=f2bf(w0.w);
                c10[4]=f2bf(w1.x); c10[5]=f2bf(w1.y); c10[6]=f2bf(w1.z); c10[7]=f2bf(w1.w);
            }
            if (k1 < 128) { c01 = *(const v8bf*)(xrow0 + k1); c11 = *(const v8bf*)(xrow1 + k1); }
            else {
                const float4* f0 = (const float4*)(arow0 + (k1 - 128));
                const float4* f1 = (const float4*)(arow1 + (k1 - 128));
                float4 u0 = f0[0], u1 = f0[1], w0 = f1[0], w1 = f1[1];
                c01[0]=f2bf(u0.x); c01[1]=f2bf(u0.y); c01[2]=f2bf(u0.z); c01[3]=f2bf(u0.w);
                c01[4]=f2bf(u1.x); c01[5]=f2bf(u1.y); c01[6]=f2bf(u1.z); c01[7]=f2bf(u1.w);
                c11[0]=f2bf(w0.x); c11[1]=f2bf(w0.y); c11[2]=f2bf(w0.z); c11[3]=f2bf(w0.w);
                c11[4]=f2bf(w1.x); c11[5]=f2bf(w1.y); c11[6]=f2bf(w1.z); c11[7]=f2bf(w1.w);
            }
            v16bf a0, a1;
            #pragma unroll
            for (int i = 0; i < 8; ++i) {
                a0[i] = c00[i]; a0[i+8] = c01[i];
                a1[i] = c10[i]; a1[i+8] = c11[i];
            }
            const bf16_t* wp = s_w1 + (size_t)ks * 8 * 512 + lane * 16;
            #pragma unroll
            for (int nt = 0; nt < 8; ++nt) {
                v16bf b = *(const v16bf*)(wp + nt * 512);
                acc0[nt] = __builtin_amdgcn_wmma_f32_16x16x32_bf16(
                    false, a0, false, b, (short)0, acc0[nt], false, false);
                acc1[nt] = __builtin_amdgcn_wmma_f32_16x16x32_bf16(
                    false, a1, false, b, (short)0, acc1[nt], false, false);
            }
        }
        #pragma unroll
        for (int nt = 0; nt < 8; ++nt) {
            int n = nt * 16 + lm;
            #pragma unroll
            for (int r = 0; r < 8; ++r) {
                int m = half * 8 + r;
                ht0[m * 136 + n] = f2bf(fmaxf(acc0[nt][r] + bn1[nt], 0.f));
                ht1[m * 136 + n] = f2bf(fmaxf(acc1[nt][r] + bn1[nt], 0.f));
            }
        }
        v8f a20[8], a21[8];
        #pragma unroll
        for (int nt = 0; nt < 8; ++nt) { a20[nt] = v8f{}; a21[nt] = v8f{}; }
        #pragma unroll
        for (int ks = 0; ks < 4; ++ks) {
            int k0 = ks * 32 + half * 8;
            v8bf c00 = *(const v8bf*)(ht0 + lm * 136 + k0);
            v8bf c01 = *(const v8bf*)(ht0 + lm * 136 + k0 + 16);
            v8bf c10 = *(const v8bf*)(ht1 + lm * 136 + k0);
            v8bf c11 = *(const v8bf*)(ht1 + lm * 136 + k0 + 16);
            v16bf a0, a1;
            #pragma unroll
            for (int i = 0; i < 8; ++i) {
                a0[i] = c00[i]; a0[i+8] = c01[i];
                a1[i] = c10[i]; a1[i+8] = c11[i];
            }
            const bf16_t* wp = s_w2 + (size_t)ks * 8 * 512 + lane * 16;
            #pragma unroll
            for (int nt = 0; nt < 8; ++nt) {
                v16bf b = *(const v16bf*)(wp + nt * 512);
                a20[nt] = __builtin_amdgcn_wmma_f32_16x16x32_bf16(
                    false, a0, false, b, (short)0, a20[nt], false, false);
                a21[nt] = __builtin_amdgcn_wmma_f32_16x16x32_bf16(
                    false, a1, false, b, (short)0, a21[nt], false, false);
            }
        }
        // tile0: residual + relu, in-place store
        #pragma unroll
        for (int nt = 0; nt < 8; ++nt) {
            int n = nt * 16 + lm;
            #pragma unroll
            for (int r = 0; r < 8; ++r) {
                int m = t0 * 16 + half * 8 + r;
                float v = a20[nt][r] + bn2[nt] + xf[(size_t)m * 128 + n];
                v = fmaxf(v, 0.f);
                xf[(size_t)m * 128 + n] = v;
                xh[(size_t)m * 128 + n] = f2bf(v);
            }
        }
        if (has1) {
            #pragma unroll
            for (int nt = 0; nt < 8; ++nt) {
                int n = nt * 16 + lm;
                #pragma unroll
                for (int r = 0; r < 8; ++r) {
                    int m = t1 * 16 + half * 8 + r;
                    float v = a21[nt][r] + bn2[nt] + xf[(size_t)m * 128 + n];
                    v = fmaxf(v, 0.f);
                    xf[(size_t)m * 128 + n] = v;
                    xh[(size_t)m * 128 + n] = f2bf(v);
                }
            }
        }
    }
}

// ---------------------------------------------------------------------------
// Softmax-over-edges reductions + scatter
// ---------------------------------------------------------------------------
__global__ void red_init_kernel(unsigned* red) {
    if (threadIdx.x == 0) { red[0] = 0u; ((float*)red)[1] = 0.f; }
}
__global__ void logit_max_kernel(const float* __restrict__ logits, int E, unsigned* red) {
    __shared__ float sm[256];
    float mx = -3.4e38f;
    for (int i = blockIdx.x * blockDim.x + threadIdx.x; i < E; i += gridDim.x * blockDim.x)
        mx = fmaxf(mx, logits[i]);
    sm[threadIdx.x] = mx; __syncthreads();
    for (int s = 128; s > 0; s >>= 1) {
        if ((int)threadIdx.x < s) sm[threadIdx.x] = fmaxf(sm[threadIdx.x], sm[threadIdx.x + s]);
        __syncthreads();
    }
    if (threadIdx.x == 0) atomicMax(red, flipf(sm[0]));
}
__global__ void logit_sum_kernel(const float* __restrict__ logits, int E, unsigned* red) {
    __shared__ float sm[256];
    float mx = unflipf(red[0]);
    float s = 0.f;
    for (int i = blockIdx.x * blockDim.x + threadIdx.x; i < E; i += gridDim.x * blockDim.x)
        s += __expf(logits[i] - mx);
    sm[threadIdx.x] = s; __syncthreads();
    for (int t = 128; t > 0; t >>= 1) {
        if ((int)threadIdx.x < t) sm[threadIdx.x] += sm[threadIdx.x + t];
        __syncthreads();
    }
    if (threadIdx.x == 0)
        __hip_atomic_fetch_add((float*)red + 1, sm[0], __ATOMIC_RELAXED, __HIP_MEMORY_SCOPE_AGENT);
}
__global__ void zero_kernel(float* p, long n) {
    for (long i = (long)blockIdx.x * blockDim.x + threadIdx.x; i < n;
         i += (long)gridDim.x * blockDim.x) p[i] = 0.f;
}
__global__ void scatter_kernel(const bf16_t* __restrict__ hbuf,
                               const float* __restrict__ logits,
                               const int* __restrict__ ei,
                               const unsigned* __restrict__ red,
                               float* __restrict__ agg, int E) {
    long total = (long)E * 16;
    float mx = unflipf(red[0]);
    float invs = 1.f / ((const float*)red)[1];
    for (long idx = (long)blockIdx.x * blockDim.x + threadIdx.x; idx < total;
         idx += (long)gridDim.x * blockDim.x) {
        int e = (int)(idx >> 4), g = (int)(idx & 15);
        float w = __expf(logits[e] - mx) * invs;
        int col = ei[(long)E + e];
        v8bf h = *(const v8bf*)(hbuf + (size_t)e * 128 + g * 8);
        float* dst = agg + (size_t)col * 128 + g * 8;
        #pragma unroll
        for (int i = 0; i < 8; ++i)
            __hip_atomic_fetch_add(dst + i, bf2f(h[i]) * w,
                                   __ATOMIC_RELAXED, __HIP_MEMORY_SCOPE_AGENT);
    }
}

// ---------------------------------------------------------------------------
// Readout: per-graph mean/max (batch = (i*B)//N is sorted & dense)
// ---------------------------------------------------------------------------
__global__ void readout_kernel(const float* __restrict__ xf, float* __restrict__ outr) {
    int g = blockIdx.x, n = threadIdx.x;  // 128 blocks x 128 threads
    int start = (int)(((long)g * NNODES + NB - 1) / NB);
    int end   = (int)(((long)(g + 1) * NNODES + NB - 1) / NB);
    float s = 0.f, mx = -3.4e38f;
    for (int i = start; i < end; ++i) {
        float v = xf[(size_t)i * 128 + n];
        s += v; mx = fmaxf(mx, v);
    }
    outr[(size_t)g * 256 + n]       = s / (float)(end - start);
    outr[(size_t)g * 256 + 128 + n] = mx;
}

// ---------------------------------------------------------------------------
// Tiny 3-token / 4-head attention, mean, concat, head-out
// ---------------------------------------------------------------------------
__global__ void mha_kernel(const float* __restrict__ q, const float* __restrict__ k,
                           const float* __restrict__ v, float* __restrict__ o) {
    __shared__ float sq[384], sk[384], sv[384], ssc[36], sa[36];
    int b = blockIdx.x, tid = threadIdx.x;
    for (int t = 0; t < 3; ++t) {
        sq[t*128+tid] = q[((size_t)b*3+t)*128+tid];
        sk[t*128+tid] = k[((size_t)b*3+t)*128+tid];
        sv[t*128+tid] = v[((size_t)b*3+t)*128+tid];
    }
    __syncthreads();
    if (tid < 36) {
        int h = tid / 9, r = tid % 9, tq = r / 3, tk = r % 3;
        float s = 0.f;
        for (int d = 0; d < 32; ++d) s += sq[tq*128+h*32+d] * sk[tk*128+h*32+d];
        ssc[tid] = s * 0.17677669529663687f;  // 1/sqrt(32)
    }
    __syncthreads();
    if (tid < 12) {
        int base = (tid / 3) * 9 + (tid % 3) * 3;
        float m = fmaxf(fmaxf(ssc[base], ssc[base+1]), ssc[base+2]);
        float e0 = __expf(ssc[base]-m), e1 = __expf(ssc[base+1]-m), e2 = __expf(ssc[base+2]-m);
        float inv = 1.f / (e0 + e1 + e2);
        sa[base] = e0*inv; sa[base+1] = e1*inv; sa[base+2] = e2*inv;
    }
    __syncthreads();
    int h = tid >> 5;
    for (int tq = 0; tq < 3; ++tq) {
        float s = sa[h*9+tq*3+0]*sv[0*128+tid]
                + sa[h*9+tq*3+1]*sv[1*128+tid]
                + sa[h*9+tq*3+2]*sv[2*128+tid];
        o[((size_t)b*3+tq)*128+tid] = s;
    }
}
__global__ void mean3_kernel(const float* __restrict__ op, float* __restrict__ out) {
    int b = blockIdx.x, n = threadIdx.x;
    out[(size_t)b*128+n] = (op[((size_t)b*3+0)*128+n] + op[((size_t)b*3+1)*128+n]
                          + op[((size_t)b*3+2)*128+n]) * (1.f/3.f);
}
__global__ void concat_kernel(const float* __restrict__ mod, const float* __restrict__ ao,
                              float* __restrict__ fused) {
    int b = blockIdx.x, n = threadIdx.x;
    fused[(size_t)b*512 + n]       = mod[(size_t)b*384 + n];
    fused[(size_t)b*512 + 128 + n] = mod[(size_t)b*384 + 128 + n];
    fused[(size_t)b*512 + 256 + n] = mod[(size_t)b*384 + 256 + n];
    fused[(size_t)b*512 + 384 + n] = ao[(size_t)b*128 + n];
}
__global__ void head2_kernel(const float* __restrict__ h1, const float* __restrict__ w2,
                             const float* __restrict__ b2, float* __restrict__ out, int t) {
    int b = blockIdx.x * blockDim.x + threadIdx.x;
    if (b >= NB) return;
    float s = 0.f;
    for (int j = 0; j < 64; ++j) s += h1[(size_t)b*64+j] * w2[j];
    out[(size_t)b*2 + t] = s + b2[0];
}

// ---------------------------------------------------------------------------
// Host-side orchestration
// ---------------------------------------------------------------------------
// input index map (setup_inputs dict order, nested insertion order)
enum {
    IN_NF = 0, IN_EI = 1, IN_EF = 2, IN_FP = 3, IN_DS = 4, IN_BATCH = 5,
    IN_NEW = 6, IN_NEB = 7, IN_EEW = 8, IN_EEB = 9,
    IN_L0 = 10, L_STRIDE = 10,  // per-layer: mw1,mb1,mw2,mb2,uw1,ub1,uw2,ub2,aw,ab
    IN_ROW = 60, IN_ROB = 61, IN_GPW = 62, IN_GPB = 63,
    IN_FPW = 64, IN_FPB = 65, IN_DPW = 66, IN_DPB = 67,
    IN_WQ = 68, IN_BQ = 69, IN_WK = 70, IN_BK = 71,
    IN_WV = 72, IN_BV = 73, IN_WO = 74, IN_BO = 75,
    IN_FM1W = 76, IN_FM1B = 77, IN_FM2W = 78, IN_FM2B = 79,
    IN_H0 = 80  // per-head: w1,b1,w2,b2
};

extern "C" void kernel_launch(void* const* d_in, const int* in_sizes, int n_in,
                              void* d_out, int out_size, void* d_ws, size_t ws_size,
                              hipStream_t stream) {
    (void)in_sizes; (void)n_in; (void)out_size; (void)ws_size;
    char* ws = (char*)d_ws;
    size_t off = 0;
    auto alloc = [&](size_t bytes) -> char* {
        char* p = ws + off;
        off += (bytes + 255) & ~(size_t)255;
        return p;
    };
    const float* nf = (const float*)d_in[IN_NF];
    const int*   ei = (const int*)d_in[IN_EI];
    const float* ef = (const float*)d_in[IN_EF];
    const float* fp = (const float*)d_in[IN_FP];
    const float* ds = (const float*)d_in[IN_DS];
    float* out = (float*)d_out;

    // big scratch
    float*  xf     = (float*)alloc((size_t)NNODES * 128 * 4);
    bf16_t* xh     = (bf16_t*)alloc((size_t)NNODES * 128 * 2);
    bf16_t* eah    = (bf16_t*)alloc((size_t)NEDGES * 128 * 2);
    bf16_t* hbuf   = (bf16_t*)alloc((size_t)NEDGES * 128 * 2);
    float*  logits = (float*)alloc((size_t)NEDGES * 4);
    float*  agg    = (float*)alloc((size_t)NNODES * 128 * 4);
    unsigned* red  = (unsigned*)alloc(256);
    float* mod     = (float*)alloc(384 * 128 * 4);
    float* dq      = (float*)alloc(384 * 128 * 4);
    float* dk      = (float*)alloc(384 * 128 * 4);
    float* dv      = (float*)alloc(384 * 128 * 4);
    float* do_     = (float*)alloc(384 * 128 * 4);
    float* dop     = (float*)alloc(384 * 128 * 4);
    float* dread   = (float*)alloc(NB * 256 * 4);
    float* dg      = (float*)alloc(NB * 128 * 4);
    float* dao     = (float*)alloc(NB * 128 * 4);
    float* dfused  = (float*)alloc(NB * 512 * 4);
    float* dz1     = (float*)alloc(NB * 256 * 4);
    float* dz      = (float*)alloc(NB * 128 * 4);
    float* dh1     = (float*)alloc(NB * 64 * 4);

    auto prepack = [&](const float* W, int K, int N, int KP) -> bf16_t* {
        size_t elems = (size_t)(KP / 32) * (N / 16) * 512;
        bf16_t* P = (bf16_t*)alloc(elems * 2);
        int blocks = (int)((elems + 255) / 256);
        prepack_kernel<<<blocks, 256, 0, stream>>>(W, P, K, N, KP);
        return P;
    };
    bf16_t* p_new = prepack((const float*)d_in[IN_NEW], 74, 128, 96);
    bf16_t* p_eew = prepack((const float*)d_in[IN_EEW], 10, 128, 32);
    bf16_t *p_mw1[5], *p_mw2[5], *p_uw1[5], *p_uw2[5];
    for (int l = 0; l < 5; ++l) {
        int b = IN_L0 + l * L_STRIDE;
        p_mw1[l] = prepack((const float*)d_in[b + 0], 384, 128, 384);
        p_mw2[l] = prepack((const float*)d_in[b + 2], 128, 128, 128);
        p_uw1[l] = prepack((const float*)d_in[b + 4], 256, 128, 256);
        p_uw2[l] = prepack((const float*)d_in[b + 6], 128, 128, 128);
    }
    bf16_t* p_row = prepack((const float*)d_in[IN_ROW], 256, 128, 256);
    bf16_t* p_gpw = prepack((const float*)d_in[IN_GPW], 128, 128, 128);
    bf16_t* p_fpw = prepack((const float*)d_in[IN_FPW], 2048, 128, 2048);
    bf16_t* p_dpw = prepack((const float*)d_in[IN_DPW], 200, 128, 224);
    bf16_t* p_wq  = prepack((const float*)d_in[IN_WQ], 128, 128, 128);
    bf16_t* p_wk  = prepack((const float*)d_in[IN_WK], 128, 128, 128);
    bf16_t* p_wv  = prepack((const float*)d_in[IN_WV], 128, 128, 128);
    bf16_t* p_wo  = prepack((const float*)d_in[IN_WO], 128, 128, 128);
    bf16_t* p_fm1 = prepack((const float*)d_in[IN_FM1W], 512, 256, 512);
    bf16_t* p_fm2 = prepack((const float*)d_in[IN_FM2W], 256, 128, 256);
    bf16_t* p_hw1[2];
    for (int t = 0; t < 2; ++t)
        p_hw1[t] = prepack((const float*)d_in[IN_H0 + t * 4], 128, 64, 128);

    // embeddings
    wmma_linear_kernel<8, 0><<<512, 128, 0, stream>>>(
        nf, 74, NNODES, 74, 96, p_new, (const float*)d_in[IN_NEB], xf, xh, 128);
    wmma_linear_kernel<8, 0><<<1024, 128, 0, stream>>>(
        ef, 10, NEDGES, 10, 32, p_eew, (const float*)d_in[IN_EEB], nullptr, eah, 128);

    // message-passing layers
    for (int l = 0; l < 5; ++l) {
        int b = IN_L0 + l * L_STRIDE;
        const float* mb1 = (const float*)d_in[b + 1];
        const float* mb2 = (const float*)d_in[b + 3];
        const float* ub1 = (const float*)d_in[b + 5];
        const float* ub2 = (const float*)d_in[b + 7];
        const float* aw  = (const float*)d_in[b + 8];
        const float* ab  = (const float*)d_in[b + 9];
        message_kernel<<<1024, 128, 166400, stream>>>(
            xh, eah, ei, p_mw1[l], p_mw2[l], mb1, mb2, aw, ab, hbuf, logits, NEDGES);
        red_init_kernel<<<1, 64, 0, stream>>>(red);
        logit_max_kernel<<<256, 256, 0, stream>>>(logits, NEDGES, red);
        logit_sum_kernel<<<256, 256, 0, stream>>>(logits, NEDGES, red);
        zero_kernel<<<4096, 256, 0, stream>>>(agg, (long)NNODES * 128);
        scatter_kernel<<<8192, 256, 0, stream>>>(hbuf, logits, ei, red, agg, NEDGES);
        update_kernel<<<512, 128, 133120, stream>>>(
            xf, xh, agg, p_uw1[l], p_uw2[l], ub1, ub2, NNODES);
    }

    // readout + fusion
    readout_kernel<<<NB, 128, 0, stream>>>(xf, dread);
    wmma_linear_kernel<8, 1><<<8, 128, 0, stream>>>(
        dread, 256, NB, 256, 256, p_row, (const float*)d_in[IN_ROB], dg, nullptr, 128);
    wmma_linear_kernel<8, 0><<<8, 128, 0, stream>>>(
        dg, 128, NB, 128, 128, p_gpw, (const float*)d_in[IN_GPB], mod + 0, nullptr, 384);
    wmma_linear_kernel<8, 0><<<8, 128, 0, stream>>>(
        fp, 2048, NB, 2048, 2048, p_fpw, (const float*)d_in[IN_FPB], mod + 128, nullptr, 384);
    wmma_linear_kernel<8, 0><<<8, 128, 0, stream>>>(
        ds, 200, NB, 200, 224, p_dpw, (const float*)d_in[IN_DPB], mod + 256, nullptr, 384);
    wmma_linear_kernel<8, 0><<<8, 128, 0, stream>>>(
        mod, 128, 384, 128, 128, p_wq, (const float*)d_in[IN_BQ], dq, nullptr, 128);
    wmma_linear_kernel<8, 0><<<8, 128, 0, stream>>>(
        mod, 128, 384, 128, 128, p_wk, (const float*)d_in[IN_BK], dk, nullptr, 128);
    wmma_linear_kernel<8, 0><<<8, 128, 0, stream>>>(
        mod, 128, 384, 128, 128, p_wv, (const float*)d_in[IN_BV], dv, nullptr, 128);
    mha_kernel<<<NB, 128, 0, stream>>>(dq, dk, dv, do_);
    wmma_linear_kernel<8, 0><<<8, 128, 0, stream>>>(
        do_, 128, 384, 128, 128, p_wo, (const float*)d_in[IN_BO], dop, nullptr, 128);
    mean3_kernel<<<NB, 128, 0, stream>>>(dop, dao);
    concat_kernel<<<NB, 128, 0, stream>>>(mod, dao, dfused);
    wmma_linear_kernel<16, 1><<<8, 128, 0, stream>>>(
        dfused, 512, NB, 512, 512, p_fm1, (const float*)d_in[IN_FM1B], dz1, nullptr, 256);
    wmma_linear_kernel<8, 1><<<8, 128, 0, stream>>>(
        dz1, 256, NB, 256, 256, p_fm2, (const float*)d_in[IN_FM2B], dz, nullptr, 128);
    for (int t = 0; t < 2; ++t) {
        int b = IN_H0 + t * 4;
        wmma_linear_kernel<4, 1><<<8, 128, 0, stream>>>(
            dz, 128, NB, 128, 128, p_hw1[t], (const float*)d_in[b + 1], dh1, nullptr, 64);
        head2_kernel<<<1, 128, 0, stream>>>(
            dh1, (const float*)d_in[b + 2], (const float*)d_in[b + 3], out, t);
    }
}
